// CompressionMLPEnsemble_33895881900618
// MI455X (gfx1250) — matrix operand order repaired and checked by reading, not compiled, not run on hardware
//
#include <hip/hip_runtime.h>

// CDNA5 / gfx1250: wave32, v_wmma_f32_16x16x32_bf16, f32 accumulate.
// Fused 3-layer ensemble MLP. Per workgroup = (member m, 128-row block).
// Weights cooperatively converted f32->bf16 into double-buffered LDS (once
// per workgroup, consumed by all 8 waves via ds_load_b128). h1/h2 in LDS.
// B-fragments and A-fragments are software-pipelined one step ahead so the
// waits before each WMMA do not drain all outstanding DS/VMEM ops.

typedef __bf16 bf16_t;
typedef __attribute__((ext_vector_type(16))) __bf16 v16bf;
typedef __attribute__((ext_vector_type(8)))  __bf16 v8bf;
typedef __attribute__((ext_vector_type(8)))  float  v8f;
typedef __attribute__((ext_vector_type(4)))  float  v4f;

#define BM        128                 // batch rows per workgroup
#define H1_STRIDE 264                 // 256 + 8 bf16 pad
#define H2_STRIDE 136                 // 128 + 8 bf16 pad
#define WSTRIDE   40                  // staged W row: 32 k + 8 pad (bf16)
#define SWBUF     (256 * WSTRIDE)     // one W staging buffer (bf16 elems)
#define W3_STRIDE 136                 // staged W3 row stride (bf16)

// ---- A fragment (16x32 bf16): raw f32 loads (issue early) + deferred cvt.
// Lane: M=lane%16, half=lane/16. K runs: [half*8..+7], [16+half*8..+7].
__device__ __forceinline__ void aload(v4f* t, const float* __restrict__ row, int kc, int half) {
  const float* p = row + kc + half * 8;
  t[0] = *(const v4f*)(p);
  t[1] = *(const v4f*)(p + 4);
  t[2] = *(const v4f*)(p + 16);
  t[3] = *(const v4f*)(p + 20);
}
__device__ __forceinline__ v16bf acvt(const v4f* t) {
  v16bf r;
#pragma unroll
  for (int i = 0; i < 4; ++i) {
    r[i]      = (__bf16)t[0][i];
    r[4 + i]  = (__bf16)t[1][i];
    r[8 + i]  = (__bf16)t[2][i];
    r[12 + i] = (__bf16)t[3][i];
  }
  return r;
}

// ---- A fragment from bf16 LDS (same run structure).
__device__ __forceinline__ v16bf fragA_lds(const bf16_t* row, int kc, int half) {
  const bf16_t* p = row + kc + half * 8;
  v8bf a = *(const v8bf*)(p);
  v8bf b = *(const v8bf*)(p + 16);
  v16bf r;
#pragma unroll
  for (int i = 0; i < 8; ++i) { r[i] = a[i]; r[8 + i] = b[i]; }
  return r;
}

// ---- B fragment (32x16, column-major) from staged bf16 LDS row.
__device__ __forceinline__ v16bf fragB_lds(const bf16_t* rowp, int half) {
  const bf16_t* p = rowp + half * 8;
  v8bf a = *(const v8bf*)(p);
  v8bf b = *(const v8bf*)(p + 16);
  v16bf r;
#pragma unroll
  for (int i = 0; i < 8; ++i) { r[i] = a[i]; r[8 + i] = b[i]; }
  return r;
}

// ---- W staging: f32 -> regs, then regs -> bf16 LDS.
__device__ __forceinline__ void wload32(v4f* t, const float* __restrict__ src) {
#pragma unroll
  for (int i = 0; i < 8; ++i) t[i] = *(const v4f*)(src + 4 * i);
}
__device__ __forceinline__ void wstore32(const v4f* t, bf16_t* dst) {
#pragma unroll
  for (int i = 0; i < 4; ++i) {
    v8bf o;
#pragma unroll
    for (int j = 0; j < 4; ++j) { o[j] = (__bf16)t[2 * i][j]; o[4 + j] = (__bf16)t[2 * i + 1][j]; }
    *(v8bf*)(dst + 8 * i) = o;
  }
}
__device__ __forceinline__ void wload16(v4f* t, const float* __restrict__ src) {
#pragma unroll
  for (int i = 0; i < 4; ++i) t[i] = *(const v4f*)(src + 4 * i);
}
__device__ __forceinline__ void wstore16(const v4f* t, bf16_t* dst) {
#pragma unroll
  for (int i = 0; i < 2; ++i) {
    v8bf o;
#pragma unroll
    for (int j = 0; j < 4; ++j) { o[j] = (__bf16)t[2 * i][j]; o[4 + j] = (__bf16)t[2 * i + 1][j]; }
    *(v8bf*)(dst + 8 * i) = o;
  }
}

__global__ void __launch_bounds__(256)
mlp_ensemble_fused(const float* __restrict__ x,
                   const float* __restrict__ W1, const float* __restrict__ b1,
                   const float* __restrict__ W2, const float* __restrict__ b2,
                   const float* __restrict__ W3, const float* __restrict__ b3,
                   float* __restrict__ out)
{
  extern __shared__ bf16_t smem[];
  bf16_t* sH1 = smem;                                   // [128][H1_STRIDE]
  bf16_t* sH2 = smem + BM * H1_STRIDE;                  // [128][H2_STRIDE]
  bf16_t* sW  = smem + BM * H1_STRIDE + BM * H2_STRIDE; // [2][256][WSTRIDE] / W3 [16][W3_STRIDE]

  const int m    = blockIdx.x >> 5;
  const int rb   = blockIdx.x & 31;
  const int tid  = threadIdx.x;
  const int w    = tid >> 5;
  const int lane = tid & 31;
  const int ln16 = lane & 15;
  const int half = lane >> 4;

  const v8f vzero = {0.f, 0.f, 0.f, 0.f, 0.f, 0.f, 0.f, 0.f};

  // =============== Layer 1: relu(X[128,1024] @ W1^T + b1) -> sH1 ===============
  const float* xrow = x + (size_t)(m * 4096 + rb * BM + w * 16 + ln16) * 1024;
  {
    v8f acc[16];
#pragma unroll
    for (int nt = 0; nt < 16; ++nt) acc[nt] = vzero;

    const float* w1row = W1 + ((size_t)m * 256 + tid) * 1024;  // stage row `tid`
    bf16_t* myrow0 = sW + tid * WSTRIDE;

    v4f wreg[8];
    wload32(wreg, w1row);            // W chunk 0
    wstore32(wreg, myrow0);

    v4f areg[4], aregN[4];
    aload(areg, xrow, 0, half);      // A chunk 0 (in flight across the barrier)
    __syncthreads();

    for (int i = 0; i < 32; ++i) {
      const int kc = i * 32;
      if (i + 1 < 32) {
        wload32(wreg, w1row + kc + 32);     // next W chunk: overlaps 16 WMMAs
        aload(aregN, xrow, kc + 32, half);  // next A chunk: overlaps 16 WMMAs
      }
      __builtin_prefetch(xrow + kc + 256, 0, 1);

      v16bf a = acvt(areg);
      const bf16_t* buf = sW + (i & 1) * SWBUF;
      v16bf bcur = fragB_lds(buf + ln16 * WSTRIDE, half);
#pragma unroll
      for (int nt = 0; nt < 16; ++nt) {
        v16bf bnxt = (nt < 15) ? fragB_lds(buf + ((nt + 1) * 16 + ln16) * WSTRIDE, half)
                               : bcur;      // rotate: next B in flight during wmma
        acc[nt] = __builtin_amdgcn_wmma_f32_16x16x32_bf16(
            false, a, false, bcur, (short)0, acc[nt], false, false);
        bcur = bnxt;
      }
      __syncthreads();
      if (i + 1 < 32) {
        wstore32(wreg, sW + ((i + 1) & 1) * SWBUF + tid * WSTRIDE);
#pragma unroll
        for (int q = 0; q < 4; ++q) areg[q] = aregN[q];
      }
      __syncthreads();
    }

#pragma unroll
    for (int nt = 0; nt < 16; ++nt) {
      float bias = b1[m * 256 + nt * 16 + ln16];
#pragma unroll
      for (int v = 0; v < 8; ++v) {
        float r = fmaxf(acc[nt][v] + bias, 0.0f);
        sH1[(w * 16 + v + 8 * half) * H1_STRIDE + nt * 16 + ln16] = (bf16_t)r;
      }
    }
  }

  // =============== Layer 2: relu(H1[128,256] @ W2^T + b2) -> sH2 ===============
  {
    v8f acc[8];
#pragma unroll
    for (int nt = 0; nt < 8; ++nt) acc[nt] = vzero;

    const int   r2    = tid & 127;
    const int   kh2   = (tid >> 7) * 16;
    const float* w2row = W2 + ((size_t)m * 128 + r2) * 256 + kh2;
    bf16_t*      myrow = sW + r2 * WSTRIDE + kh2;

    v4f wreg[4];
    wload16(wreg, w2row);            // chunk 0
    wstore16(wreg, myrow);
    __syncthreads();                 // also publishes sH1 epilogue stores

    const bf16_t* h1row = sH1 + (w * 16 + ln16) * H1_STRIDE;
    for (int i = 0; i < 8; ++i) {
      const int kc = i * 32;
      if (i + 1 < 8) wload16(wreg, w2row + kc + 32);

      v16bf a = fragA_lds(h1row, kc, half);
      const bf16_t* buf = sW + (i & 1) * SWBUF;
      v16bf bcur = fragB_lds(buf + ln16 * WSTRIDE, half);
#pragma unroll
      for (int nt = 0; nt < 8; ++nt) {
        v16bf bnxt = (nt < 7) ? fragB_lds(buf + ((nt + 1) * 16 + ln16) * WSTRIDE, half)
                              : bcur;
        acc[nt] = __builtin_amdgcn_wmma_f32_16x16x32_bf16(
            false, a, false, bcur, (short)0, acc[nt], false, false);
        bcur = bnxt;
      }
      __syncthreads();
      if (i + 1 < 8) wstore16(wreg, sW + ((i + 1) & 1) * SWBUF + r2 * WSTRIDE + kh2);
      __syncthreads();
    }

#pragma unroll
    for (int nt = 0; nt < 8; ++nt) {
      float bias = b2[m * 128 + nt * 16 + ln16];
#pragma unroll
      for (int v = 0; v < 8; ++v) {
        float r = fmaxf(acc[nt][v] + bias, 0.0f);
        sH2[(w * 16 + v + 8 * half) * H2_STRIDE + nt * 16 + ln16] = (bf16_t)r;
      }
    }
  }

  // =============== Layer 3: tanh(H2[128,128] @ W3^T + b3) -> out ===============
  {
    // stage all of W3 (16x128) as bf16 into sW: thread -> (row = tid/16, 8 k's)
    const int r3 = tid >> 4;
    const int k3 = (tid & 15) * 8;
    const float* src = W3 + ((size_t)m * 16 + r3) * 128 + k3;
    v4f a0 = *(const v4f*)(src);
    v4f a1 = *(const v4f*)(src + 4);
    v8bf o3;
#pragma unroll
    for (int j = 0; j < 4; ++j) { o3[j] = (__bf16)a0[j]; o3[4 + j] = (__bf16)a1[j]; }
    *(v8bf*)(sW + r3 * W3_STRIDE + k3) = o3;
    __syncthreads();                 // publishes sH2 + staged W3

    v8f acc = vzero;
    const bf16_t* h2row = sH2 + (w * 16 + ln16) * H2_STRIDE;
    const bf16_t* w3row = sW + ln16 * W3_STRIDE;
#pragma unroll
    for (int kc = 0; kc < 128; kc += 32) {
      v16bf a  = fragA_lds(h2row, kc, half);
      v16bf bb = fragB_lds(w3row + kc, half);
      acc = __builtin_amdgcn_wmma_f32_16x16x32_bf16(
          false, a, false, bb, (short)0, acc, false, false);
    }

    float bias = b3[m * 16 + ln16];
#pragma unroll
    for (int v = 0; v < 8; ++v) {
      float r = tanhf(acc[v] + bias);
      int grow = rb * BM + w * 16 + v + 8 * half;
      out[((size_t)m * 4096 + grow) * 16 + ln16] = r;
    }
  }
}

extern "C" void kernel_launch(void* const* d_in, const int* in_sizes, int n_in,
                              void* d_out, int out_size, void* d_ws, size_t ws_size,
                              hipStream_t stream) {
  (void)in_sizes; (void)n_in; (void)out_size; (void)d_ws; (void)ws_size;
  const float* x  = (const float*)d_in[0];
  const float* W1 = (const float*)d_in[1];
  const float* b1 = (const float*)d_in[2];
  const float* W2 = (const float*)d_in[3];
  const float* b2 = (const float*)d_in[4];
  const float* W3 = (const float*)d_in[5];
  const float* b3 = (const float*)d_in[6];
  float* out = (float*)d_out;

  size_t shmem = (size_t)(BM * H1_STRIDE + BM * H2_STRIDE + 2 * SWBUF) * sizeof(bf16_t); // 143360 B
  dim3 grid(16 * 32);
  dim3 block(256);
  mlp_ensemble_fused<<<grid, block, shmem, stream>>>(x, W1, b1, W2, b2, W3, b3, out);
}
